// GraphAttention_64287070486899
// MI455X (gfx1250) — matrix operand order repaired and checked by reading, not compiled, not run on hardware
//
#include <hip/hip_runtime.h>
#include <hip/hip_bf16.h>
#include <math.h>

typedef __attribute__((ext_vector_type(16))) __bf16 v16bf;
typedef __attribute__((ext_vector_type(8)))  float  v8f;
typedef int v4i __attribute__((vector_size(16)));    // matches builtin param pointee

#if defined(__gfx1250__) && __has_builtin(__builtin_amdgcn_global_load_async_to_lds_b128)
#define HAVE_ASYNC 1
#else
#define HAVE_ASYNC 0
#endif

union Frag { v16bf v; uint4 q[2]; };

__device__ __forceinline__ unsigned short f2bf(float x) {
    unsigned int u = __float_as_uint(x);
    unsigned int r = (u + 0x7FFFu + ((u >> 16) & 1u)) >> 16;   // RNE truncate to bf16
    return (unsigned short)r;
}

#define N_NODES 2048
#define FDIM    128
#define HEADS   4
#define ROWS    32      // output rows per block
#define MT      64      // m (K) tile per iteration
#define NT      (N_NODES / MT)
#define PPITCH  72      // bf16 elems per p row (144B, 16B aligned)
#define XPITCH  72      // bf16 elems per xT row

__device__ __forceinline__ v16bf load_afrag(const unsigned short* p) {
    Frag f; f.q[0] = *(const uint4*)(p); f.q[1] = *(const uint4*)(p + 16); return f.v;
}
__device__ __forceinline__ v16bf load_bfrag(const unsigned short* p) {
    Frag f; f.q[0] = *(const uint4*)(p); f.q[1] = *(const uint4*)(p + 16); return f.v;
}

// ---------------- kernel 1: a_self / a_neigh = X . attn^T -------------------
__global__ void gat_coef(const float* __restrict__ X,
                         const float* __restrict__ attn_self,
                         const float* __restrict__ attn_neigh,
                         float* __restrict__ a_self,
                         float* __restrict__ a_neigh,
                         int BN) {
    int gid = blockIdx.x * blockDim.x + threadIdx.x;   // over B*N*H, h fastest
    if (gid >= BN * HEADS) return;
    int h    = gid & (HEADS - 1);
    int node = gid >> 2;
    const float4* xr = (const float4*)(X + (size_t)node * FDIM);
    const float4* ws = (const float4*)(attn_self  + h * FDIM);
    const float4* wn = (const float4*)(attn_neigh + h * FDIM);
    float ss = 0.f, sn = 0.f;
    #pragma unroll 8
    for (int i = 0; i < FDIM / 4; ++i) {
        float4 xv = xr[i], sv = ws[i], nv = wn[i];
        ss += xv.x * sv.x + xv.y * sv.y + xv.z * sv.z + xv.w * sv.w;
        sn += xv.x * nv.x + xv.y * nv.y + xv.z * nv.z + xv.w * nv.w;
    }
    int b = node / N_NODES, n = node % N_NODES;
    a_self [((size_t)b * HEADS + h) * N_NODES + n] = ss;
    a_neigh[((size_t)b * HEADS + h) * N_NODES + n] = sn;
}

// ------------- kernel 2: X [B,N,F] f32 -> X^T [B,F,N] bf16 ------------------
__global__ void gat_tr(const float* __restrict__ X, unsigned short* __restrict__ XbfT) {
    __shared__ unsigned short tile[32][72];            // [f][n], padded
    int bz = blockIdx.x;
    int b  = bz >> 7;
    int rem = bz & 127;
    int n0 = (rem >> 2) * 64, f0 = (rem & 3) * 32;
    int t = threadIdx.x;
    #pragma unroll
    for (int i = 0; i < 8; ++i) {                      // 64n x 32f, f fastest (coalesced)
        int idx = t + i * 256;
        int n = idx >> 5, f = idx & 31;
        tile[f][n] = f2bf(X[((size_t)b * N_NODES + n0 + n) * FDIM + f0 + f]);
    }
    __syncthreads();
    #pragma unroll
    for (int i = 0; i < 8; ++i) {                      // write rows of X^T (coalesced)
        int idx = t + i * 256;
        int f = idx >> 6, n = idx & 63;
        XbfT[((size_t)b * FDIM + f0 + f) * N_NODES + n0 + n] = tile[f][n];
    }
}

// ---------------- kernel 3: fused masked-softmax attention ------------------
__shared__ unsigned short xT_s[2][FDIM * XPITCH];
__shared__ unsigned short pL_s[2][HEADS * ROWS * PPITCH];

__device__ __forceinline__ void stage_x_tile(const unsigned short* __restrict__ XbfT,
                                             unsigned short* xbuf, int b, int m0, int t) {
    #pragma unroll
    for (int k = 0; k < 4; ++k) {                      // 128 rows x 8 chunks of 16B
        int c = t + k * 256;
        int f = c >> 3, o = c & 7;
        const unsigned short* g = XbfT + ((size_t)(b * FDIM + f) * N_NODES + m0) + o * 8;
        unsigned short* l = xbuf + f * XPITCH + o * 8;
#if HAVE_ASYNC
        __builtin_amdgcn_global_load_async_to_lds_b128(
            (__attribute__((address_space(1))) v4i*)g,
            (__attribute__((address_space(3))) v4i*)l, 0, 0);
#else
        *(uint4*)l = *(const uint4*)g;
#endif
    }
}

__global__ void __launch_bounds__(256, 1) gat_main(
    const unsigned short* __restrict__ XbfT, const float* __restrict__ A,
    const float* __restrict__ a_self, const float* __restrict__ a_neigh,
    float* __restrict__ out)
{
    __shared__ float anL[HEADS * N_NODES];             // a_neigh for this batch (32KB)
    __shared__ float asel[HEADS * ROWS];
    __shared__ float psum[256];
    __shared__ float rowsum[HEADS * ROWS];

    const int t    = threadIdx.x;
    const int b    = blockIdx.x >> 6;                  // N/ROWS = 64 row-tiles per batch
    const int row0 = (blockIdx.x & 63) * ROWS;

    // p-compute mapping: 2 threads per (head,row), each covers 32 of 64 m's
    const int pairIdx = t >> 1;                        // 0..127 = h*32 + r
    const int ph   = pairIdx >> 5;
    const int pr   = pairIdx & 31;
    const int psub = t & 1;

    // wave mapping: wave w -> (head, row-half); accumulates 16 rows x 128 f
    const int w     = t >> 5;
    const int lane  = t & 31;
    const int hw    = w >> 1;
    const int rh    = w & 1;
    const int colL  = lane & 15;
    const int halfL = lane >> 4;

    for (int i = t; i < HEADS * N_NODES; i += 256)
        anL[i] = a_neigh[(size_t)b * HEADS * N_NODES + i];
    if (t < HEADS * ROWS)
        asel[t] = a_self[((size_t)b * HEADS + (t >> 5)) * N_NODES + row0 + (t & 31)];

    stage_x_tile(XbfT, &xT_s[0][0], b, 0, t);          // prefetch tile 0

    v8f zero = {0.f, 0.f, 0.f, 0.f, 0.f, 0.f, 0.f, 0.f};
    v8f acc[8];
    #pragma unroll
    for (int j = 0; j < 8; ++j) acc[j] = zero;
    float lsum = 0.f;

    __syncthreads();
    const float a_s = asel[pairIdx];
    const float* Arow = A + ((size_t)b * N_NODES + row0 + pr) * N_NODES;

    int buf = 0;
    for (int it = 0; it < NT; ++it, buf ^= 1) {
        const int m0 = it * MT;
        // ---- compute unnormalized probs p into pL_s[buf] ----
        {
            unsigned short* prow = &pL_s[buf][pairIdx * PPITCH + psub * 32];
            const float4* a4 = (const float4*)(Arow + m0 + psub * 32);
            const float4* n4 = (const float4*)(&anL[ph * N_NODES + m0 + psub * 32]);
            #pragma unroll
            for (int q = 0; q < 8; ++q) {
                float4 av = a4[q], nv = n4[q];
                float e0 = a_s + nv.x; e0 = (e0 >= 0.f) ? e0 : 0.2f * e0;
                float e1 = a_s + nv.y; e1 = (e1 >= 0.f) ? e1 : 0.2f * e1;
                float e2 = a_s + nv.z; e2 = (e2 >= 0.f) ? e2 : 0.2f * e2;
                float e3 = a_s + nv.w; e3 = (e3 >= 0.f) ? e3 : 0.2f * e3;
                float p0 = (av.x == 0.f) ? 0.f : __expf(e0);
                float p1 = (av.y == 0.f) ? 0.f : __expf(e1);
                float p2 = (av.z == 0.f) ? 0.f : __expf(e2);
                float p3 = (av.w == 0.f) ? 0.f : __expf(e3);
                lsum += p0 + p1 + p2 + p3;
                ushort4 st = { f2bf(p0), f2bf(p1), f2bf(p2), f2bf(p3) };
                *(ushort4*)(prow + q * 4) = st;
            }
        }
#if HAVE_ASYNC
#if __has_builtin(__builtin_amdgcn_s_wait_asynccnt)
        __builtin_amdgcn_s_wait_asynccnt(0);
#else
        asm volatile("s_wait_asynccnt 0" ::: "memory");
#endif
#endif
        __syncthreads();
        // prefetch next X tile into the other buffer (safe: everyone passed barrier)
        if (it + 1 < NT)
            stage_x_tile(XbfT, &xT_s[buf ^ 1][0], b, m0 + MT, t);

        // ---- WMMA: acc[16 rows x 128 f] += P[16 x 64] * X[64 x 128] ----
        const unsigned short* pA0 = &pL_s[buf][(hw * ROWS + rh * 16 + colL) * PPITCH];
        #pragma unroll
        for (int ks = 0; ks < 2; ++ks) {
            v16bf fa = load_afrag(pA0 + ks * 32 + halfL * 8);
            #pragma unroll
            for (int j = 0; j < 8; ++j) {
                v16bf fb = load_bfrag(&xT_s[buf][(j * 16 + colL) * XPITCH + ks * 32 + halfL * 8]);
                acc[j] = __builtin_amdgcn_wmma_f32_16x16x32_bf16(
                    false, fa, false, fb, (short)0, acc[j], false, false);
            }
        }
    }

    // ---- reduce row sums (2 partials per (h,r)) ----
    psum[t] = lsum;
    __syncthreads();
    if (t < HEADS * ROWS)
        rowsum[t] = psum[2 * t] + psum[2 * t + 1];
    __syncthreads();

    // ---- epilogue: normalize, tanh, write concat-head output ----
    #pragma unroll
    for (int j = 0; j < 8; ++j) {
        #pragma unroll
        for (int v = 0; v < 8; ++v) {
            int row = rh * 16 + v + 8 * halfL;          // C layout: VGPR v -> row
            float s = rowsum[hw * ROWS + row];
            float val = tanhf(acc[j][v] / s);
            out[((size_t)b * N_NODES + row0 + row) * (HEADS * FDIM)
                + hw * FDIM + j * 16 + colL] = val;
        }
    }
}

extern "C" void kernel_launch(void* const* d_in, const int* in_sizes, int n_in,
                              void* d_out, int out_size, void* d_ws, size_t ws_size,
                              hipStream_t stream) {
    const float* X    = (const float*)d_in[0];
    const float* A    = (const float*)d_in[1];
    const float* as_w = (const float*)d_in[2];
    const float* an_w = (const float*)d_in[3];
    float* out = (float*)d_out;

    int B  = in_sizes[0] / (N_NODES * FDIM);
    int BN = B * N_NODES;

    float* a_self  = (float*)d_ws;
    float* a_neigh = a_self + (size_t)BN * HEADS;
    unsigned short* XbfT = (unsigned short*)(a_neigh + (size_t)BN * HEADS);

    int total = BN * HEADS;
    hipLaunchKernelGGL(gat_coef, dim3((total + 255) / 256), dim3(256), 0, stream,
                       X, as_w, an_w, a_self, a_neigh, BN);
    hipLaunchKernelGGL(gat_tr, dim3(B * 128), dim3(256), 0, stream, X, XbfT);
    hipLaunchKernelGGL(gat_main, dim3(B * (N_NODES / ROWS)), dim3(256), 0, stream,
                       XbfT, A, a_self, a_neigh, out);
}
